// Estimator_29351806501040
// MI455X (gfx1250) — matrix operand order repaired and checked by reading, not compiled, not run on hardware
//
#include <hip/hip_runtime.h>

// MI455X / gfx1250, wave32. Conv layers = implicit GEMM on
// v_wmma_f32_16x16x32_f16, K enumerated tap-major (tap constant per 32-chunk)
// so im2col staging is ~2 VALU/element with immediate-offset global loads.
// A chunks streamed with global_load_async_to_lds_b128 (ASYNCcnt), LDS
// double-buffered; fragments stored in WMMA register order -> ds_load_b128.

typedef __attribute__((ext_vector_type(16))) _Float16 v16h;
typedef __attribute__((ext_vector_type(8)))  float    v8f;
typedef _Float16 half_t;

#define BATCH 4
#define Hh 512
#define Ww 512
#define HW (Hh * Ww)
#define OC 64
#define NPIX (BATCH * Hh * Ww)
#define MAXP 184

__device__ __forceinline__ int refl(int i, int n) {
  i = (i < 0) ? -i : i;
  i = (i >= n) ? (2 * n - 2 - i) : i;
  return i;
}

// One 16-byte async global->LDS copy (gfx1250, tracked by ASYNCcnt).
__device__ __forceinline__ void async_copy_b128(unsigned lds_byte_addr,
                                                const void* gaddr) {
  asm volatile("global_load_async_to_lds_b128 %0, %1, off"
               :: "v"(lds_byte_addr), "v"(gaddr) : "memory");
}
__device__ __forceinline__ void wait_async0() {
  asm volatile("s_wait_asynccnt 0" ::: "memory");
}

// ---------------- blurs ----------------

__global__ void copy_ch0(const float* __restrict__ nx, half_t* __restrict__ xf) {
  int idx = blockIdx.x * blockDim.x + threadIdx.x;
  if (idx >= NPIX) return;
  int x = idx % Ww, y = (idx / Ww) % Hh, b = idx / (Ww * Hh);
  xf[(((size_t)b * 5 + 0) * Hh + y) * Ww + x] = (half_t)nx[idx];
}

// Horizontal: one block per image row; row + reflected halo staged in LDS,
// so the tap loop is ds_load + fma only. 2 pixels per thread.
__global__ __launch_bounds__(256) void blur_h_lds(
    const float* __restrict__ src, const float* __restrict__ gk,
    int taps, float* __restrict__ dst) {
  __shared__ float srow[Ww + 2 * MAXP];
  const int p = (taps - 1) >> 1;
  const int row = blockIdx.x;                // b*Hh + y
  const float* g = src + (size_t)row * Ww;
  const int tot = Ww + 2 * p;
  for (int j = threadIdx.x; j < tot; j += 256)
    srow[j] = g[refl(j - p, Ww)];
  __syncthreads();
  const int x0 = threadIdx.x, x1 = threadIdx.x + 256;
  float a0 = 0.f, a1 = 0.f;
  for (int t = 0; t < taps; ++t) {
    float gv = gk[t];
    a0 += gv * srow[x0 + t];
    a1 += gv * srow[x1 + t];
  }
  dst[(size_t)row * Ww + x0] = a0;
  dst[(size_t)row * Ww + x1] = a1;
}

// Vertical: 2 adjacent pixels per thread share reflect/address math and gk.
__global__ void blur_v2(const float* __restrict__ tmp, const float* __restrict__ gk,
                        int taps, half_t* __restrict__ xf, int ch) {
  int idx = blockIdx.x * blockDim.x + threadIdx.x;
  if (idx >= NPIX / 2) return;
  int xh = idx % (Ww / 2), y = (idx / (Ww / 2)) % Hh, b = idx / ((Ww / 2) * Hh);
  int x = xh * 2;
  int p = (taps - 1) >> 1;
  float a0 = 0.f, a1 = 0.f;
  for (int t = 0; t < taps; ++t) {
    int iy = refl(y + t - p, Hh);
    const float* rp = tmp + ((size_t)b * Hh + iy) * Ww + x;
    float gv = gk[t];
    a0 += gv * rp[0];
    a1 += gv * rp[1];
  }
  half_t* o = xf + (((size_t)b * 5 + ch) * Hh + y) * Ww + x;
  o[0] = (half_t)a0;
  o[1] = (half_t)a1;
}

// --------- weight repack: tap-major K, WMMA-A fragment order, zero pad ---------
// kc -> tap t = kc/cpt, channel base cb = (kc%cpt)*32.
// In-chunk K contiguous per lane half (same as B staging): K = e + (lane>=16?16:0).

__global__ void wcvt(const float* __restrict__ w, half_t* __restrict__ Awf,
                     int Cin, int cpt, int nelem) {
  int idx = blockIdx.x * blockDim.x + threadIdx.x;
  if (idx >= nelem) return;
  int kc = idx >> 11;
  int r2 = idx & 2047;
  int mtile = r2 >> 9;
  int r3 = r2 & 511;
  int lane = r3 >> 4;
  int e = r3 & 15;
  int t  = kc / cpt;
  int cb = (kc - t * cpt) * 32;
  int r = t / 5, s = t - r * 5;
  int K = e + ((lane & 16) ? 16 : 0);
  int c = cb + K;
  int m = mtile * 16 + (lane & 15);
  Awf[idx] = (c < Cin) ? (half_t)w[((size_t)m * Cin + c) * 25 + r * 5 + s]
                       : (half_t)0.f;
}

// ---------------- implicit-GEMM conv + bias + PReLU ----------------
// 256 threads = 8 waves; tile 64 out-ch x 128 px. Wave w: M-tile w&3,
// N-tiles 4*(w>>2)..+3. 25*CPT chunks of K=32 (tap constant per chunk).

template <int WRITE_F32, int CPT>
__global__ __launch_bounds__(256) void conv_wmma(
    const half_t* __restrict__ inY, int CinY,
    const half_t* __restrict__ inX, int CinX,
    const half_t* __restrict__ Awf,
    const float* __restrict__ bias, const float* __restrict__ alpha,
    half_t* __restrict__ outH, float* __restrict__ outF)
{
  __shared__ __align__(16) half_t As[2][4 * 32 * 16];   // 2 x 4KB
  __shared__ __align__(16) half_t Bs[2][8 * 32 * 16];   // 2 x 8KB

  const int tid    = threadIdx.x;
  const int lane   = tid & 31;
  const int wave   = tid >> 5;
  const int lanelo = lane & 15;
  const int lanehi = (lane >> 4) & 1;

  const int x0 = blockIdx.x * 128;
  const int y  = blockIdx.y;
  const int b  = blockIdx.z;

  const int Cin = CinY + CinX;
  const int nch = 25 * CPT;

  const int mtile  = wave & 3;
  const int ntbase = (wave >> 2) * 4;

  // per-thread staging coordinates
  const int stN     = (tid >> 5) * 16 + lanelo;  // pixel within 128-tile
  const int stKhi   = lanehi ? 16 : 0;           // B frag K offset
  const int ix0base = x0 + stN - 2;

  v8f acc[4] = {v8f{}, v8f{}, v8f{}, v8f{}};

  auto stageA = [&](int kc, int buf) {
    unsigned dst = (unsigned)(size_t)&As[buf][0] + (unsigned)tid * 16u;
    const char* src = (const char*)(Awf + (size_t)kc * 2048) + (size_t)tid * 16;
    async_copy_b128(dst, src);
  };

  auto stageB = [&](int kc, int buf) {
    const int t  = kc / CPT;              // compile-time CPT -> mul-shift
    const int cb = (kc - t * CPT) * 32;
    const int r  = t / 5, s = t - r * 5;
    const int iy = refl(y + r - 2, Hh);
    const int ix = refl(ix0base + s, Ww);
    const half_t* base;
    int cmax;
    if (cb < CinY) {                      // uniform per chunk (CinY multiple of 32)
      base = inY + (((size_t)(b * CinY + cb)) * Hh + iy) * Ww + ix;
      cmax = CinY - cb;
    } else {
      base = inX + (((size_t)(b * CinX + (cb - CinY))) * Hh + iy) * Ww + ix;
      cmax = Cin - cb;
    }
    base += (size_t)stKhi * HW;
    const int rem = cmax - stKhi;         // uniform (cmax in {64,32,5})
    v16h v;
    if (rem >= 16) {
#pragma unroll
      for (int e = 0; e < 16; ++e) v[e] = base[(size_t)e * HW];  // imm offsets
    } else {
#pragma unroll
      for (int e = 0; e < 16; ++e)
        v[e] = (e < rem) ? base[(size_t)e * HW] : (half_t)0.f;
    }
    *(v16h*)&Bs[buf][tid * 16] = v;       // tid*16 == ntile*512 + lane*16
  };

  // prologue
  stageA(0, 0);
  stageB(0, 0);
  wait_async0();
  __syncthreads();

  for (int i = 0; i < nch; ++i) {
    const int buf = i & 1;
    if (i + 1 < nch) {                    // overlap next-chunk staging
      stageA(i + 1, buf ^ 1);
      stageB(i + 1, buf ^ 1);
      if (i + 2 < nch)
        __builtin_prefetch(Awf + (size_t)(i + 2) * 2048, 0, 0);
    }

    v16h a = *(const v16h*)&As[buf][mtile * 512 + lane * 16];
#pragma unroll
    for (int j = 0; j < 4; ++j) {
      v16h bf = *(const v16h*)&Bs[buf][(ntbase + j) * 512 + lane * 16];
      acc[j] = __builtin_amdgcn_wmma_f32_16x16x32_f16(false, a, false, bf,
                                                      (short)0, acc[j],
                                                      false, false);
    }

    wait_async0();
    __syncthreads();
  }

  // epilogue: bias + PReLU, store
  const float al = alpha[0];
#pragma unroll
  for (int r = 0; r < 8; ++r) {
    const int m = mtile * 16 + r + lanehi * 8;
    const float bv = bias[m];
    const size_t rowbase = (((size_t)b * OC + m) * Hh + y) * Ww + x0;
#pragma unroll
    for (int j = 0; j < 4; ++j) {
      float v = acc[j][r] + bv;
      v = (v >= 0.f) ? v : al * v;
      size_t o = rowbase + (ntbase + j) * 16 + lanelo;
      if (WRITE_F32) {
        __builtin_nontemporal_store(v, &outF[o]);  // 256MB stream, bypass L2
      } else {
        outH[o] = (half_t)v;
      }
    }
  }
}

// ---------------- host orchestration ----------------

extern "C" void kernel_launch(void* const* d_in, const int* in_sizes, int n_in,
                              void* d_out, int out_size, void* d_ws, size_t ws_size,
                              hipStream_t stream) {
  (void)n_in; (void)out_size; (void)ws_size;

  const float* nx = (const float*)d_in[0];
  const float* gk[4]; int taps[4];
  for (int i = 0; i < 4; ++i) { gk[i] = (const float*)d_in[1 + i]; taps[i] = in_sizes[1 + i]; }
  const float* wgt[4]; const float* bia[4]; const float* alp[4];
  for (int i = 0; i < 4; ++i) {
    wgt[i] = (const float*)d_in[5 + 3 * i];
    bia[i] = (const float*)d_in[6 + 3 * i];
    alp[i] = (const float*)d_in[7 + 3 * i];
  }
  float* out = (float*)d_out;

  // Workspace carve-up (all offsets 16B-aligned).
  char* ws = (char*)d_ws;
  size_t off = 0;
  float*  tmp = (float*)(ws + off);  off += (size_t)NPIX * sizeof(float);       // 4 MB
  half_t* xf  = (half_t*)(ws + off); off += (size_t)BATCH * 5 * HW * 2;         // 10 MB
  half_t* yA  = (half_t*)(ws + off); off += (size_t)BATCH * OC * HW * 2;        // 128 MB
  half_t* yB  = (half_t*)(ws + off); off += (size_t)BATCH * OC * HW * 2;        // 128 MB
  half_t* Awf = (half_t*)(ws + off); off += (size_t)75 * 2048 * 2;              // 300 KB

  const int TB = 256;

  // Feature stack x = [nx, blur0..blur3] in f16.
  copy_ch0<<<dim3((NPIX + TB - 1) / TB), TB, 0, stream>>>(nx, xf);
  for (int i = 0; i < 4; ++i) {
    blur_h_lds<<<dim3(BATCH * Hh), TB, 0, stream>>>(nx, gk[i], taps[i], tmp);
    blur_v2<<<dim3((NPIX / 2 + TB - 1) / TB), TB, 0, stream>>>(tmp, gk[i], taps[i], xf, i + 1);
  }

  dim3 cgrid(Ww / 128, Hh, BATCH);

  // Layer 0: x(5) -> yA   (CPT = 1)
  {
    int nelem = 25 * 1 * 2048;
    wcvt<<<(nelem + TB - 1) / TB, TB, 0, stream>>>(wgt[0], Awf, 5, 1, nelem);
    conv_wmma<0, 1><<<cgrid, TB, 0, stream>>>(nullptr, 0, xf, 5, Awf,
                                              bia[0], alp[0], yA, nullptr);
  }
  // Layers 1-3: Cin = 69 -> CPT = 3
  {
    int nelem = 25 * 3 * 2048;
    wcvt<<<(nelem + TB - 1) / TB, TB, 0, stream>>>(wgt[1], Awf, 69, 3, nelem);
    conv_wmma<0, 3><<<cgrid, TB, 0, stream>>>(yA, OC, xf, 5, Awf,
                                              bia[1], alp[1], yB, nullptr);

    wcvt<<<(nelem + TB - 1) / TB, TB, 0, stream>>>(wgt[2], Awf, 69, 3, nelem);
    conv_wmma<0, 3><<<cgrid, TB, 0, stream>>>(yB, OC, xf, 5, Awf,
                                              bia[2], alp[2], yA, nullptr);

    wcvt<<<(nelem + TB - 1) / TB, TB, 0, stream>>>(wgt[3], Awf, 69, 3, nelem);
    conv_wmma<1, 3><<<cgrid, TB, 0, stream>>>(yA, OC, xf, 5, Awf,
                                              bia[3], alp[3], nullptr, out);
  }
}
